// MKAARMACell_10642928960162
// MI455X (gfx1250) — matrix operand order repaired and checked by reading, not compiled, not run on hardware
//
#include <hip/hip_runtime.h>
#include <math.h>

typedef _Float16 v8h  __attribute__((ext_vector_type(8)));
typedef _Float16 v16h __attribute__((ext_vector_type(16)));
typedef float    v8f  __attribute__((ext_vector_type(8)));

constexpr int Bn = 512, Mn = 16, Dn = 128, Pn = 16, Tn = 4096;
constexpr int KAUG = 160;  // 128 (state) + 16 (phi) + 16 zero-pad -> 5 k-steps of 32

// ---------------- staging kernels ----------------

// X[b, 0:128]=state, X[b,128:144]=phi, X[b,144:160]=0   (f16)
__global__ void build_x(const float* __restrict__ state, const float* __restrict__ phi,
                        _Float16* __restrict__ Xh) {
  int idx = blockIdx.x * 256 + threadIdx.x;
  if (idx >= Bn * KAUG) return;
  int b = idx / KAUG, c = idx % KAUG;
  float v = 0.f;
  if (c < Dn)            v = state[b * Dn + c];
  else if (c < Dn + Pn)  v = phi[b * Pn + (c - Dn)];
  Xh[idx] = (_Float16)v;
}

// W[m, h, 0:128]=A[m,:,h] (transposed), W[m,h,128:144]=B_in[m,:,h], rest 0   (f16)
__global__ void build_w(const float* __restrict__ A, const float* __restrict__ Bin,
                        _Float16* __restrict__ Wh) {
  int idx = blockIdx.x * 256 + threadIdx.x;
  if (idx >= Mn * Dn * KAUG) return;
  int m = idx / (Dn * KAUG);
  int rem = idx % (Dn * KAUG);
  int h = rem / KAUG, d = rem % KAUG;
  float v = 0.f;
  if (d < Dn)            v = A[((size_t)m * Dn + d) * Dn + h];
  else if (d < Dn + Pn)  v = Bin[((size_t)m * Pn + (d - Dn)) * Dn + h];
  Wh[idx] = (_Float16)v;
}

// per-row: f16 copy of trajectories + ||t||^2  (one wave per row of 128)
__global__ void build_traj(const float* __restrict__ traj, _Float16* __restrict__ Th,
                           float* __restrict__ tsq) {
  int wave = (blockIdx.x * 256 + threadIdx.x) >> 5;
  int lane = threadIdx.x & 31;
  if (wave >= Tn) return;
  float4 v = *(const float4*)(traj + (size_t)wave * Dn + lane * 4);
  _Float16* o = Th + (size_t)wave * Dn + lane * 4;
  o[0] = (_Float16)v.x; o[1] = (_Float16)v.y; o[2] = (_Float16)v.z; o[3] = (_Float16)v.w;
  float ss = v.x * v.x + v.y * v.y + v.z * v.z + v.w * v.w;
  for (int off = 16; off; off >>= 1) ss += __shfl_xor(ss, off, 32);
  if (lane == 0) tsq[wave] = ss;
}

// ||new_state row||^2 over D=128 (one wave per bm row), reads f32 new_state from d_out tail
__global__ void row_sq(const float* __restrict__ ns, float* __restrict__ ssq) {
  int wave = (blockIdx.x * 256 + threadIdx.x) >> 5;
  int lane = threadIdx.x & 31;
  if (wave >= Bn * Mn) return;
  float4 v = *(const float4*)(ns + (size_t)wave * Dn + lane * 4);
  float ss = v.x * v.x + v.y * v.y + v.z * v.z + v.w * v.w;
  for (int off = 16; off; off >>= 1) ss += __shfl_xor(ss, off, 32);
  if (lane == 0) ssq[wave] = ss;
}

// ---------------- stage 1: new_state = tanh(X @ W_m^T), K=160 via 5 WMMA ----------------
__global__ __launch_bounds__(256) void stage1(const _Float16* __restrict__ Xh,
                                              const _Float16* __restrict__ Wh,
                                              float* __restrict__ new_state,
                                              _Float16* __restrict__ Sh) {
  int waveId = (blockIdx.x * 256 + threadIdx.x) >> 5;  // 0..4095 = 32 btiles * 16 m * 8 htiles
  int lane = threadIdx.x & 31;
  int hi = lane >> 4, n = lane & 15;
  int htile = waveId & 7;
  int m = (waveId >> 3) & 15;
  int b0 = (waveId >> 7) * 16;
  int h0 = htile * 16;

  const _Float16* xrow = Xh + (size_t)(b0 + n) * KAUG;            // A-operand: row = lane&15
  const _Float16* wcol = Wh + ((size_t)m * Dn + h0 + n) * KAUG;   // B-operand: col = lane&15

  v8f acc = {};
#pragma unroll
  for (int k0 = 0; k0 < KAUG; k0 += 32) {
    v8h alo = *(const v8h*)(xrow + k0 + hi * 8);
    v8h ahi = *(const v8h*)(xrow + k0 + 16 + hi * 8);
    v16h a = __builtin_shufflevector(alo, ahi, 0,1,2,3,4,5,6,7,8,9,10,11,12,13,14,15);
    v8h blo = *(const v8h*)(wcol + k0 + hi * 8);
    v8h bhi = *(const v8h*)(wcol + k0 + 16 + hi * 8);
    v16h bf = __builtin_shufflevector(blo, bhi, 0,1,2,3,4,5,6,7,8,9,10,11,12,13,14,15);
    acc = __builtin_amdgcn_wmma_f32_16x16x32_f16(false, a, false, bf, (short)0, acc,
                                                 false, false);
  }
#pragma unroll
  for (int r = 0; r < 8; ++r) {
    int b = b0 + r + hi * 8;                 // C row = r + 8*hi
    float v = tanhf(acc[r]);
    new_state[((size_t)b * Mn + m) * Dn + h0 + n] = v;
    Sh[((size_t)b * Mn + m) * Dn + h0 + n] = (_Float16)v;
  }
}

// ---------------- stage 2: sim = exp(-max(||s||^2+||t||^2-2 s.t, 0)) ----------------
// 512 bm-tiles x 256 t-tiles; one wave per 8 consecutive t-tiles (A-frags reused in regs)
__global__ __launch_bounds__(256) void stage2(const _Float16* __restrict__ Sh,
                                              const _Float16* __restrict__ Th,
                                              const float* __restrict__ ssq,
                                              const float* __restrict__ tsq,
                                              float* __restrict__ out) {
  int waveId = (blockIdx.x * 256 + threadIdx.x) >> 5;  // 0..16383
  int lane = threadIdx.x & 31;
  int hi = lane >> 4, n = lane & 15;
  int bmTile = waveId >> 5;     // 512
  int tGroup = waveId & 31;     // 32 groups of 8 t-tiles
  int bm0 = bmTile * 16;

  const _Float16* srow = Sh + (size_t)(bm0 + n) * Dn;
  v16h afr[4];
#pragma unroll
  for (int kk = 0; kk < 4; ++kk) {
    int k0 = kk * 32;
    v8h alo = *(const v8h*)(srow + k0 + hi * 8);
    v8h ahi = *(const v8h*)(srow + k0 + 16 + hi * 8);
    afr[kk] = __builtin_shufflevector(alo, ahi, 0,1,2,3,4,5,6,7,8,9,10,11,12,13,14,15);
  }
  float4 sa = *(const float4*)(ssq + bm0 + hi * 8);
  float4 sb = *(const float4*)(ssq + bm0 + hi * 8 + 4);
  float sq[8] = {sa.x, sa.y, sa.z, sa.w, sb.x, sb.y, sb.z, sb.w};

#pragma unroll
  for (int i = 0; i < 8; ++i) {
    int t0 = (tGroup * 8 + i) * 16;
    const _Float16* trow = Th + (size_t)(t0 + n) * Dn + hi * 8;  // B col = lane&15
    v8f acc = {};
#pragma unroll
    for (int kk = 0; kk < 4; ++kk) {
      v8h blo = *(const v8h*)(trow + kk * 32);
      v8h bhi = *(const v8h*)(trow + kk * 32 + 16);
      v16h bf = __builtin_shufflevector(blo, bhi, 0,1,2,3,4,5,6,7,8,9,10,11,12,13,14,15);
      acc = __builtin_amdgcn_wmma_f32_16x16x32_f16(false, afr[kk], false, bf, (short)0,
                                                   acc, false, false);
    }
    float tq = tsq[t0 + n];
#pragma unroll
    for (int r = 0; r < 8; ++r) {
      float d = sq[r] + tq - 2.f * acc[r];
      d = fmaxf(d, 0.f);
      out[(size_t)(bm0 + r + hi * 8) * Tn + t0 + n] = __expf(-d);  // GAMMA = 1
    }
  }
}

// ---------------- launch ----------------
extern "C" void kernel_launch(void* const* d_in, const int* in_sizes, int n_in,
                              void* d_out, int out_size, void* d_ws, size_t ws_size,
                              hipStream_t stream) {
  (void)in_sizes; (void)n_in; (void)out_size; (void)ws_size;
  const float* phi   = (const float*)d_in[0];
  const float* state = (const float*)d_in[1];
  const float* traj  = (const float*)d_in[2];
  const float* A     = (const float*)d_in[3];
  const float* Bin   = (const float*)d_in[4];

  float* out = (float*)d_out;
  float* new_state = out + (size_t)Bn * Mn * Tn;  // second tuple output

  char* ws = (char*)d_ws;
  _Float16* Xh = (_Float16*)(ws);              //   512*160*2 = 163840
  _Float16* Wh = (_Float16*)(ws + 163840);     // 16*128*160*2 = 655360
  _Float16* Th = (_Float16*)(ws + 819200);     // 4096*128*2 = 1048576
  _Float16* Sh = (_Float16*)(ws + 1867776);    // 8192*128*2 = 2097152
  float*   tsq = (float*)(ws + 3964928);       // 4096*4
  float*   ssq = (float*)(ws + 3981312);       // 8192*4  (end ~3.83 MB)

  build_x<<<(Bn * KAUG + 255) / 256, 256, 0, stream>>>(state, phi, Xh);
  build_w<<<(Mn * Dn * KAUG + 255) / 256, 256, 0, stream>>>(A, Bin, Wh);
  build_traj<<<Tn / 8, 256, 0, stream>>>(traj, Th, tsq);
  stage1<<<512, 256, 0, stream>>>(Xh, Wh, new_state, Sh);   // 4096 waves
  row_sq<<<(Bn * Mn) / 8, 256, 0, stream>>>(new_state, ssq);
  stage2<<<2048, 256, 0, stream>>>(Sh, Th, ssq, tsq, out);  // 16384 waves
}
